// GraphVAE_5403068858879
// MI455X (gfx1250) — matrix operand order repaired and checked by reading, not compiled, not run on hardware
//
#include <hip/hip_runtime.h>
#include <hip/hip_bf16.h>

typedef _Float16 half_t;
typedef __attribute__((ext_vector_type(8)))  half_t v8h;
typedef __attribute__((ext_vector_type(16))) half_t v16h;
typedef __attribute__((ext_vector_type(8)))  float  v8f;

#define NEG_SLOPE 0.2f

// -------------------- helpers --------------------

__device__ __forceinline__ void atomic_max_float(float* addr, float val) {
  // sign-magnitude ordered atomic max for IEEE f32 (no NaNs in this workload)
  if (val >= 0.0f) atomicMax((int*)addr, __float_as_int(val));
  else             atomicMin((unsigned int*)addr, __float_as_uint(val));
}

// -------------------- conversion kernels --------------------

__global__ void k_cast_half(const float* __restrict__ src, half_t* __restrict__ dst, long n) {
  long i = (long)blockIdx.x * blockDim.x + threadIdx.x;
  if (i < n) dst[i] = (half_t)src[i];
}

// W [K, C] row-major f32  ->  Wt [C, K] row-major f16   (i.e. B^T for WMMA B-fragments)
__global__ void k_transpose_cast(const float* __restrict__ W, half_t* __restrict__ Wt, int K, int C) {
  int i = blockIdx.x * blockDim.x + threadIdx.x;
  if (i >= K * C) return;
  int k = i / C, c = i % C;
  Wt[(long)c * K + k] = (half_t)W[i];
}

__global__ void k_fill(float* __restrict__ p, float v, long n) {
  long i = (long)blockIdx.x * blockDim.x + threadIdx.x;
  if (i < n) p[i] = v;
}

// -------------------- WMMA GEMM --------------------
// C[M,Ncols] (f32, optional f16 copy) = A[M,K](f16, row major) * Bt[Ncols,K](f16, row major == B^T)
// One wave computes one 16x16 tile; K in steps of 32 via v_wmma_f32_16x16x32_f16.
__global__ __launch_bounds__(256)
void k_gemm_wmma(const half_t* __restrict__ A, const half_t* __restrict__ Bt,
                 float* __restrict__ C, half_t* __restrict__ C16,
                 const float* __restrict__ bias, int M, int K, int Ncols, int relu) {
  const int lane   = threadIdx.x & 31;
  const int laneLo = lane & 15;
  const int laneHi = lane >> 4;
  const int wavesPerBlock = blockDim.x >> 5;
  const int waveId  = blockIdx.x * wavesPerBlock + (threadIdx.x >> 5);
  const int nWaves  = gridDim.x * wavesPerBlock;
  const int tilesN  = Ncols >> 4;
  const int nTiles  = (M >> 4) * tilesN;

  for (int t = waveId; t < nTiles; t += nWaves) {
    const int row0 = (t / tilesN) << 4;
    const int col0 = (t % tilesN) << 4;
    const int arow = row0 + laneLo;
    const int bcol = col0 + laneLo;
    v8f acc = {};
    for (int kk = 0; kk < K; kk += 32) {
      // A fragment: lane holds (row=arow), k in [kk+8*laneHi, +8) and [kk+16+8*laneHi, +8)
      const half_t* ap = A + (long)arow * K + kk + laneHi * 8;
      v8h alo = *(const v8h*)(ap);
      v8h ahi = *(const v8h*)(ap + 16);
      // B fragment: lane holds (col=bcol), k in [kk+8*laneHi, +8) and [kk+16+8*laneHi, +8)
      const half_t* bp = Bt + (long)bcol * K + kk + laneHi * 8;
      v8h blo = *(const v8h*)(bp);
      v8h bhi = *(const v8h*)(bp + 16);
      v16h a, b;
#pragma unroll
      for (int i = 0; i < 8; ++i) { a[i] = alo[i]; a[i + 8] = ahi[i];
                                    b[i] = blo[i]; b[i + 8] = bhi[i]; }
      acc = __builtin_amdgcn_wmma_f32_16x16x32_f16(false, a, false, b,
                                                   (short)0, acc, false, false);
    }
#pragma unroll
    for (int r = 0; r < 8; ++r) {
      const int row = row0 + r + laneHi * 8;   // VGPR r: lanes0-15 M=r, lanes16-31 M=r+8
      const int col = col0 + laneLo;
      float v = acc[r];
      if (bias) v += bias[col];
      if (relu) v = fmaxf(v, 0.0f);
      const long o = (long)row * Ncols + col;
      C[o] = v;
      if (C16) C16[o] = (half_t)v;
    }
  }
}

// -------------------- GAT attention kernels --------------------

// per (node, head): s = <h[n,head,:], a[head,:]>
__global__ void k_scores(const float* __restrict__ h, const float* __restrict__ a_src,
                         const float* __restrict__ a_dst, float* __restrict__ s_src,
                         float* __restrict__ s_dst, int n, int H, int C) {
  int idx = blockIdx.x * blockDim.x + threadIdx.x;
  if (idx >= n * H) return;
  int nn = idx / H, hh = idx % H;
  const float* hp = h + (long)nn * H * C + (long)hh * C;
  const float* as = a_src + hh * C;
  const float* ad = a_dst + hh * C;
  float ss = 0.f, sd = 0.f;
  for (int c = 0; c < C; ++c) { float v = hp[c]; ss += v * as[c]; sd += v * ad[c]; }
  s_src[idx] = ss; s_dst[idx] = sd;
}

__device__ __forceinline__ void edge_sd(const int* ei, int e, int E, int& s, int& d) {
  if (e < E) { s = ei[e]; d = ei[E + e]; } else { s = d = e - E; } // appended self loops
}

__global__ void k_edge_max(const float* __restrict__ ssrc, const float* __restrict__ sdst,
                           const int* __restrict__ ei, float* __restrict__ m,
                           float* __restrict__ ebuf, int EN, int H, int E) {
  int idx = blockIdx.x * blockDim.x + threadIdx.x;
  if (idx >= EN * H) return;
  int e = idx / H, h = idx % H, s, d;
  edge_sd(ei, e, E, s, d);
  float v = ssrc[s * H + h] + sdst[d * H + h];
  v = (v > 0.f) ? v : NEG_SLOPE * v;
  ebuf[idx] = v;
  atomic_max_float(&m[d * H + h], v);
}

__global__ void k_edge_exp(float* __restrict__ ebuf, const float* __restrict__ m,
                           float* __restrict__ denom, const int* __restrict__ ei,
                           int EN, int H, int E) {
  int idx = blockIdx.x * blockDim.x + threadIdx.x;
  if (idx >= EN * H) return;
  int e = idx / H, h = idx % H, s, d;
  edge_sd(ei, e, E, s, d);
  float v = expf(ebuf[idx] - m[d * H + h]);
  ebuf[idx] = v;
  atomicAdd(&denom[d * H + h], v);
}

// one thread per (edge, head, channel): 32 consecutive lanes cover 32 consecutive
// channels of the SAME destination row -> coalesced gather of h[src] and coalesced
// per-cacheline global_atomic_add_f32 into out[dst].
__global__ void k_edge_scatter(const float* __restrict__ ebuf, const float* __restrict__ denom,
                               const int* __restrict__ ei, const float* __restrict__ hfeat,
                               float* __restrict__ outacc, long total, int H, int C, int E) {
  long idx = (long)blockIdx.x * blockDim.x + threadIdx.x;
  if (idx >= total) return;                 // total = EN * H * C
  int c  = (int)(idx % C);
  int eh = (int)(idx / C);                  // e*H + h
  int e  = eh / H, h = eh % H, s, d;
  edge_sd(ei, e, E, s, d);
  float alpha = ebuf[eh] / (denom[d * H + h] + 1e-16f);
  float v = alpha * hfeat[(long)s * H * C + (long)h * C + c];
  atomicAdd(&outacc[(long)d * H * C + (long)h * C + c], v);
}

__global__ void k_epilogue(const float* __restrict__ acc, const float* __restrict__ bias,
                           float* __restrict__ outF, half_t* __restrict__ outH,
                           int relu, long total, int HC) {
  long idx = (long)blockIdx.x * blockDim.x + threadIdx.x;
  if (idx >= total) return;
  float v = acc[idx] + bias[(int)(idx % HC)];
  if (relu) v = fmaxf(v, 0.f);
  if (outF) outF[idx] = v;
  if (outH) outH[idx] = (half_t)v;
}

// -------------------- decoder kernels --------------------

__global__ void k_reparam(const float* __restrict__ mu, const float* __restrict__ lv,
                          const float* __restrict__ eps, float* __restrict__ z,
                          half_t* __restrict__ z16, long n) {
  long i = (long)blockIdx.x * blockDim.x + threadIdx.x;
  if (i >= n) return;
  float v = mu[i] + eps[i] * expf(0.5f * lv[i]);
  z[i] = v; z16[i] = (half_t)v;
}

__global__ void k_edge_out(const float* __restrict__ zd, const int* __restrict__ ei,
                           const float* __restrict__ We, const float* __restrict__ be,
                           float* __restrict__ out, int E, int HID_) {
  int e = blockIdx.x * blockDim.x + threadIdx.x;
  if (e >= E) return;
  const float* a = zd + (long)ei[e] * HID_;
  const float* b = zd + (long)ei[E + e] * HID_;
  float acc = be[0];
  for (int c = 0; c < HID_; ++c) acc += a[c] * b[c] * We[c];
  out[e] = acc;
}

// -------------------- host --------------------

extern "C" void kernel_launch(void* const* d_in, const int* in_sizes, int n_in,
                              void* d_out, int out_size, void* d_ws, size_t ws_size,
                              hipStream_t stream) {
  constexpr int N = 50000, E = 400000, IN = 64, HID = 64, H = 4, LAT = 32;
  constexpr int HC = H * HID;        // 256
  constexpr int EN = E + N;          // with self loops

  const float* x      = (const float*)d_in[0];
  const int*   ei     = (const int*)  d_in[1];
  const float* eps    = (const float*)d_in[3];
  const float* W1     = (const float*)d_in[4];
  const float* asrc1  = (const float*)d_in[5];
  const float* adst1  = (const float*)d_in[6];
  const float* b1     = (const float*)d_in[7];
  const float* W2     = (const float*)d_in[8];
  const float* asrc2  = (const float*)d_in[9];
  const float* adst2  = (const float*)d_in[10];
  const float* b2     = (const float*)d_in[11];
  const float* Wmu    = (const float*)d_in[12];
  const float* asrcmu = (const float*)d_in[13];
  const float* adstmu = (const float*)d_in[14];
  const float* bmu    = (const float*)d_in[15];
  const float* Wlv    = (const float*)d_in[16];
  const float* asrclv = (const float*)d_in[17];
  const float* adstlv = (const float*)d_in[18];
  const float* blv    = (const float*)d_in[19];
  const float* Wfc    = (const float*)d_in[20];
  const float* bfc    = (const float*)d_in[21];
  const float* Wnode  = (const float*)d_in[22];
  const float* bnode  = (const float*)d_in[23];
  const float* Wedge  = (const float*)d_in[24];
  const float* bedge  = (const float*)d_in[25];

  // ---- workspace layout ----
  char* ws = (char*)d_ws;
  size_t off = 0;
  auto take = [&](size_t bytes) -> char* {
    char* p = ws + off; off = (off + bytes + 255) & ~(size_t)255; return p;
  };
  float*  tmpA  = (float*) take((size_t)N * HC * 4);   // pre-attention h
  float*  tmpB  = (float*) take((size_t)N * HC * 4);   // attention accumulator
  half_t* a16   = (half_t*)take((size_t)N * HC * 2);   // f16 activations (x/h1/h2/z)
  half_t* zd16  = (half_t*)take((size_t)N * HID * 2);
  float*  zd    = (float*) take((size_t)N * HID * 4);
  float*  ssrc  = (float*) take((size_t)N * H * 4);
  float*  sdst  = (float*) take((size_t)N * H * 4);
  float*  mbuf  = (float*) take((size_t)N * H * 4);
  float*  dbuf  = (float*) take((size_t)N * H * 4);
  float*  ebuf  = (float*) take((size_t)EN * H * 4);
  half_t* w1t   = (half_t*)take((size_t)HC * IN * 2);
  half_t* w2t   = (half_t*)take((size_t)HC * HC * 2);
  half_t* wmut  = (half_t*)take((size_t)LAT * HC * 2);
  half_t* wlvt  = (half_t*)take((size_t)LAT * HC * 2);
  half_t* wfct  = (half_t*)take((size_t)HID * LAT * 2);
  half_t* wnt   = (half_t*)take((size_t)IN * HID * 2);

  float* out_mu   = (float*)d_out;
  float* out_lv   = out_mu  + (size_t)N * LAT;
  float* out_z    = out_lv  + (size_t)N * LAT;
  float* out_node = out_z   + (size_t)N * LAT;
  float* out_edge = out_node + (size_t)N * IN;

  const int T = 256;
  auto nb = [&](long n) { return (int)((n + T - 1) / T); };

  // ---- prep: casts + weight transposes ----
  k_cast_half<<<nb((long)N * IN), T, 0, stream>>>(x, a16, (long)N * IN);
  k_transpose_cast<<<nb(IN * HC),  T, 0, stream>>>(W1,   w1t,  IN,  HC);
  k_transpose_cast<<<nb(HC * HC),  T, 0, stream>>>(W2,   w2t,  HC,  HC);
  k_transpose_cast<<<nb(HC * LAT), T, 0, stream>>>(Wmu,  wmut, HC,  LAT);
  k_transpose_cast<<<nb(HC * LAT), T, 0, stream>>>(Wlv,  wlvt, HC,  LAT);
  k_transpose_cast<<<nb(LAT * HID),T, 0, stream>>>(Wfc,  wfct, LAT, HID);
  k_transpose_cast<<<nb(HID * IN), T, 0, stream>>>(Wnode,wnt,  HID, IN);

  // ---- one GAT layer ----
  auto gat = [&](const half_t* Ain, int K, const half_t* Wt, const float* as,
                 const float* ad, const float* bias, int Hh, int C,
                 float* outF, half_t* outH16, int relu) {
    const int HCl   = Hh * C;
    const int tiles = (N / 16) * (HCl / 16);
    k_gemm_wmma<<<(tiles + 7) / 8, 256, 0, stream>>>(Ain, Wt, tmpA, (half_t*)nullptr,
                                                     (const float*)nullptr, N, K, HCl, 0);
    k_scores<<<nb((long)N * Hh), T, 0, stream>>>(tmpA, as, ad, ssrc, sdst, N, Hh, C);
    k_fill<<<nb((long)N * Hh),  T, 0, stream>>>(mbuf, -3.402823466e38f, (long)N * Hh);
    k_fill<<<nb((long)N * Hh),  T, 0, stream>>>(dbuf, 0.0f, (long)N * Hh);
    k_fill<<<nb((long)N * HCl), T, 0, stream>>>(tmpB, 0.0f, (long)N * HCl);
    k_edge_max<<<nb((long)EN * Hh), T, 0, stream>>>(ssrc, sdst, ei, mbuf, ebuf, EN, Hh, E);
    k_edge_exp<<<nb((long)EN * Hh), T, 0, stream>>>(ebuf, mbuf, dbuf, ei, EN, Hh, E);
    const long scatter_total = (long)EN * Hh * C;
    k_edge_scatter<<<nb(scatter_total), T, 0, stream>>>(ebuf, dbuf, ei, tmpA, tmpB,
                                                        scatter_total, Hh, C, E);
    k_epilogue<<<nb((long)N * HCl), T, 0, stream>>>(tmpB, bias, outF, outH16, relu,
                                                    (long)N * HCl, HCl);
  };

  // ---- encoder ----
  gat(a16, IN, w1t, asrc1, adst1, b1, H, HID, nullptr, a16, 1);   // h1 -> a16 (f16)
  gat(a16, HC, w2t, asrc2, adst2, b2, H, HID, nullptr, a16, 1);   // h2 -> a16 (f16)
  gat(a16, HC, wmut, asrcmu, adstmu, bmu, 1, LAT, out_mu, nullptr, 0);
  gat(a16, HC, wlvt, asrclv, adstlv, blv, 1, LAT, out_lv, nullptr, 0);

  // ---- reparameterize: z = mu + eps*exp(0.5*logvar) ----
  k_reparam<<<nb((long)N * LAT), T, 0, stream>>>(out_mu, out_lv, eps, out_z, a16, (long)N * LAT);

  // ---- decoder ----
  {
    const int tiles_fc = (N / 16) * (HID / 16);
    k_gemm_wmma<<<(tiles_fc + 7) / 8, 256, 0, stream>>>(a16, wfct, zd, zd16, bfc,
                                                        N, LAT, HID, 1);   // zd = relu(z@Wfc+b)
    const int tiles_nd = (N / 16) * (IN / 16);
    k_gemm_wmma<<<(tiles_nd + 7) / 8, 256, 0, stream>>>(zd16, wnt, out_node, (half_t*)nullptr,
                                                        bnode, N, HID, IN, 0);
  }
  k_edge_out<<<nb(E), T, 0, stream>>>(zd, ei, Wedge, bedge, out_edge, E, HID);
}
